// Quantum_cell_29119878266886
// MI455X (gfx1250) — matrix operand order repaired and checked by reading, not compiled, not run on hardware
//
#include <hip/hip_runtime.h>

// ---------------- problem constants ----------------
constexpr int B = 512, T = 512, DRIVE = 5, INP = 64, OUT = 64, CS = 32, TRANS = 4;
constexpr int ROWS = 16;            // batch rows per workgroup (== WMMA M)
constexpr int NBLK = B / ROWS;      // 32 workgroups
constexpr int NTH  = 256;           // 8 wave32 per workgroup

// ---------------- types for WMMA ----------------
typedef __attribute__((ext_vector_type(16))) __bf16 bf16x16;
typedef __attribute__((ext_vector_type(8)))  float  f32x8;

union FragU { bf16x16 v; uint4 q[2]; };

__device__ __forceinline__ unsigned short f2bf(float x) {
    unsigned u = __float_as_uint(x);
    unsigned r = u + 0x7FFFu + ((u >> 16) & 1u);   // round-to-nearest-even
    return (unsigned short)(r >> 16);
}
__device__ __forceinline__ float bflo(unsigned v) { return __uint_as_float(v << 16); }
__device__ __forceinline__ float bfhi(unsigned v) { return __uint_as_float(v & 0xFFFF0000u); }
__device__ __forceinline__ unsigned packbf(float a, float b) {
    return (unsigned)f2bf(a) | ((unsigned)f2bf(b) << 16);
}

// branch-free tanh: 1 - 2/(e^{2x}+1), via v_exp_f32 (2^x) + v_rcp_f32.
// clamp keeps exp2 argument in range; |err| ~1e-7, far below bf16 matmul noise.
__device__ __forceinline__ float tanh_fast(float x) {
    float xc = fminf(fmaxf(x, -9.0f), 9.0f);
    float e2x = __builtin_amdgcn_exp2f(xc * 2.8853900817779268f);  // e^{2x}
    return 1.0f - 2.0f * __builtin_amdgcn_rcpf(e2x + 1.0f);
}

// A fragment (16x32 bf16, contiguous k = 16*half + s, matching B packing) from row-major LDS [16][rowStride]
__device__ __forceinline__ bf16x16 load_afrag(const unsigned short* base, int rowStride,
                                              int k0, int lane) {
    int m = lane & 15, half = (lane >> 4) & 1;
    const unsigned short* p = base + m * rowStride + k0 + half * 16;
    FragU f;
    f.q[0] = *reinterpret_cast<const uint4*>(p);
    f.q[1] = *reinterpret_cast<const uint4*>(p + 8);
    return f.v;
}

// B fragment pre-swizzled in LDS as [frag][lane][8 dwords] -> two b128 loads
__device__ __forceinline__ bf16x16 load_bfrag(const unsigned* frags, int fragIdx, int lane) {
    const unsigned* p = frags + ((fragIdx * 32 + lane) << 3);
    FragU f;
    f.q[0] = *reinterpret_cast<const uint4*>(p);
    f.q[1] = *reinterpret_cast<const uint4*>(p + 4);
    return f.v;
}

__device__ __forceinline__ f32x8 wmma_bf16(bf16x16 a, bf16x16 b, f32x8 c) {
    return __builtin_amdgcn_wmma_f32_16x16x32_bf16(false, a, false, b, (short)0, c, false, false);
}

// ---------------- LDS layout (~316 KB of the 320 KB WGP LDS) ----------------
struct SMem {
    alignas(16) unsigned       wud[64 * 2 * 32 * 8];   // 128 KB  W_ud b-frags (64 ntiles x 2 ktiles)
    alignas(16) unsigned       wut[64 * 2 * 32 * 8];   // 128 KB  W_ut b-frags
    alignas(16) unsigned       wtd[4 * 2 * 32 * 8];    //   8 KB  W_td b-frags
    alignas(16) unsigned       wts[4 * 32 * 8];        //   4 KB  W_ts b-frags (K=32)
    alignas(16) unsigned short U[ROWS * 1024];         //  32 KB  tanh(U) staging, bf16
    alignas(16) unsigned short inp[ROWS * INP];        //   2 KB
    alignas(16) unsigned short tg[ROWS * OUT];         //   2 KB
    alignas(16) float          cf[ROWS * CS];          //   2 KB  cell state f32
    alignas(16) unsigned short cbf[ROWS * CS];         //   1 KB  cell state bf16 (A-matrix)
    alignas(16) float          cnew[ROWS * CS];        //   2 KB
    alignas(16) float          budut[1024];            //   4 KB  b_ud + b_ut
    alignas(16) float          btdts[OUT];             //  b_td + b_ts
    alignas(16) float          win[DRIVE * INP];
    alignas(16) float          bin[INP];
    alignas(16) float          woutT[TRANS * OUT];     //  W_out transposed: [o][q]
    alignas(16) float          bout[TRANS];
    alignas(16) float          drv[ROWS * DRIVE];
};

__global__ __launch_bounds__(NTH, 1) void quantum_cell_kernel(
    const float* __restrict__ drive, const float* __restrict__ cell,
    const float* __restrict__ W_in,  const float* __restrict__ b_in,
    const float* __restrict__ W_td,  const float* __restrict__ b_td,
    const float* __restrict__ W_ts,  const float* __restrict__ b_ts,
    const float* __restrict__ W_ud,  const float* __restrict__ b_ud,
    const float* __restrict__ W_ut,  const float* __restrict__ b_ut,
    const float* __restrict__ W_out, const float* __restrict__ b_out,
    float* __restrict__ out_trans,   float* __restrict__ out_cell) {

    __shared__ SMem sm;
    const int tid  = threadIdx.x;
    const int lane = tid & 31;
    const int wave = tid >> 5;
    const int r0   = blockIdx.x * ROWS;

    // ---------- one-time setup: pack weights into bf16 WMMA fragments ----------
    for (int idx = tid; idx < 64 * 2 * 32 * 8; idx += NTH) {
        int r = idx & 7, ln = (idx >> 3) & 31, f = idx >> 8;
        int ntile = f >> 1, ktile = f & 1;
        int n = ntile * 16 + (ln & 15);
        int k = ktile * 32 + ((ln >> 4) & 1) * 16 + r * 2;
        sm.wud[idx] = packbf(W_ud[k * 1024 + n], W_ud[(k + 1) * 1024 + n]);
        sm.wut[idx] = packbf(W_ut[k * 1024 + n], W_ut[(k + 1) * 1024 + n]);
    }
    for (int idx = tid; idx < 8 * 32 * 8; idx += NTH) {
        int r = idx & 7, ln = (idx >> 3) & 31, f = idx >> 8;
        int ntile = f >> 1, ktile = f & 1;
        int n = ntile * 16 + (ln & 15);
        int k = ktile * 32 + ((ln >> 4) & 1) * 16 + r * 2;
        sm.wtd[idx] = packbf(W_td[k * OUT + n], W_td[(k + 1) * OUT + n]);
    }
    for (int idx = tid; idx < 4 * 32 * 8; idx += NTH) {
        int r = idx & 7, ln = (idx >> 3) & 31, ntile = idx >> 8;
        int n = ntile * 16 + (ln & 15);
        int k = ((ln >> 4) & 1) * 16 + r * 2;
        sm.wts[idx] = packbf(W_ts[k * OUT + n], W_ts[(k + 1) * OUT + n]);
    }
    for (int idx = tid; idx < 1024; idx += NTH) sm.budut[idx] = b_ud[idx] + b_ut[idx];
    if (tid < OUT)          sm.btdts[tid] = b_td[tid] + b_ts[tid];
    if (tid < DRIVE * INP)  sm.win[tid]   = W_in[tid];
    if (tid < INP)          sm.bin[tid]   = b_in[tid];
    if (tid < OUT * TRANS) {                       // transpose W_out -> [o][q]
        int q = tid >> 2, o = tid & 3;
        sm.woutT[o * OUT + q] = W_out[q * TRANS + o];
    }
    if (tid < TRANS)        sm.bout[tid] = b_out[tid];
    for (int idx = tid; idx < ROWS * CS; idx += NTH) {
        int m = idx >> 5, j = idx & 31;
        float v = cell[(r0 + m) * CS + j];
        sm.cf[idx]  = v;
        sm.cbf[idx] = f2bf(v);
    }
    // drive for t=0
    if (tid < ROWS * DRIVE) {
        int m = tid / DRIVE, k = tid % DRIVE;
        sm.drv[tid] = drive[((r0 + m) * T + 0) * DRIVE + k];
    }
    __syncthreads();

    // ---------- the sequential recurrence ----------
    for (int t = 0; t < T; ++t) {
        // issue next step's drive loads NOW (into registers); commit to LDS at end
        // of the step, so global latency overlaps the whole step's compute.
        float drv_next = 0.0f;
        if (tid < ROWS * DRIVE && t + 1 < T) {
            int m = tid / DRIVE, k = tid % DRIVE;
            drv_next = drive[((r0 + m) * T + (t + 1)) * DRIVE + k];
        }

        // inp = d @ W_in + b_in   (16x64, trivial VALU)
        #pragma unroll
        for (int e = 0; e < (ROWS * INP) / NTH; ++e) {
            int idx = e * NTH + tid, m = idx >> 6, n = idx & 63;
            float v = sm.bin[n];
            #pragma unroll
            for (int k = 0; k < DRIVE; ++k) v += sm.drv[m * DRIVE + k] * sm.win[k * INP + n];
            sm.inp[idx] = f2bf(v);
        }
        __syncthreads();

        // Tg = tanh(inp@W_td + c@W_ts + b)  : 4 N-tiles on waves 0..3, 3 WMMAs each
        if (wave < 4) {
            bf16x16 a0 = load_afrag(sm.inp, INP, 0,  lane);
            bf16x16 a1 = load_afrag(sm.inp, INP, 32, lane);
            bf16x16 ac = load_afrag(sm.cbf, CS,  0,  lane);
            f32x8 acc = {};
            acc = wmma_bf16(a0, load_bfrag(sm.wtd, wave * 2 + 0, lane), acc);
            acc = wmma_bf16(a1, load_bfrag(sm.wtd, wave * 2 + 1, lane), acc);
            acc = wmma_bf16(ac, load_bfrag(sm.wts, wave,         lane), acc);
            int half = lane >> 4, n = wave * 16 + (lane & 15);
            #pragma unroll
            for (int r = 0; r < 8; ++r) {
                int m = r + half * 8;
                sm.tg[m * OUT + n] = f2bf(tanh_fast(acc[r] + sm.btdts[n]));
            }
        }
        __syncthreads();

        // U = tanh(inp@W_ud + Tg@W_ut + b) : 64 N-tiles, 8 per wave, 4 WMMAs each
        {
            bf16x16 a0 = load_afrag(sm.inp, INP, 0,  lane);
            bf16x16 a1 = load_afrag(sm.inp, INP, 32, lane);
            bf16x16 g0 = load_afrag(sm.tg,  OUT, 0,  lane);
            bf16x16 g1 = load_afrag(sm.tg,  OUT, 32, lane);
            int half = lane >> 4;
            #pragma unroll
            for (int tt = 0; tt < 8; ++tt) {
                int ntile = wave * 8 + tt;
                f32x8 acc = {};
                acc = wmma_bf16(a0, load_bfrag(sm.wud, ntile * 2 + 0, lane), acc);
                acc = wmma_bf16(a1, load_bfrag(sm.wud, ntile * 2 + 1, lane), acc);
                acc = wmma_bf16(g0, load_bfrag(sm.wut, ntile * 2 + 0, lane), acc);
                acc = wmma_bf16(g1, load_bfrag(sm.wut, ntile * 2 + 1, lane), acc);
                int n = ntile * 16 + (lane & 15);
                float bias = sm.budut[n];
                #pragma unroll
                for (int r = 0; r < 8; ++r) {
                    int m = r + half * 8;
                    sm.U[m * 1024 + n] = f2bf(tanh_fast(acc[r] + bias));
                }
            }
        }
        __syncthreads();

        // c_new[m][i] = sum_j U[m][i*32+j] * c[m][j]  (vectorized b128 LDS reads)
        #pragma unroll
        for (int e = 0; e < (ROWS * CS) / NTH; ++e) {
            int idx = e * NTH + tid, m = idx >> 5, i = idx & 31;
            const uint4*  Uq = reinterpret_cast<const uint4*>(&sm.U[m * 1024 + i * CS]);
            const float4* cq = reinterpret_cast<const float4*>(&sm.cf[m * CS]);
            float s = 0.f;
            #pragma unroll
            for (int q = 0; q < 4; ++q) {
                uint4  u  = Uq[q];
                float4 c0 = cq[2 * q], c1 = cq[2 * q + 1];
                s += bflo(u.x) * c0.x + bfhi(u.x) * c0.y;
                s += bflo(u.y) * c0.z + bfhi(u.y) * c0.w;
                s += bflo(u.z) * c1.x + bfhi(u.z) * c1.y;
                s += bflo(u.w) * c1.z + bfhi(u.w) * c1.w;
            }
            sm.cnew[idx] = s;
        }
        // trans = Tg @ W_out + b_out  (16x4, vectorized, written straight to global)
        if (tid < ROWS * TRANS) {
            int m = tid >> 2, o = tid & 3;
            const uint4*  Tq = reinterpret_cast<const uint4*>(&sm.tg[m * OUT]);
            const float4* Wq = reinterpret_cast<const float4*>(&sm.woutT[o * OUT]);
            float s = sm.bout[o];
            #pragma unroll
            for (int q = 0; q < 8; ++q) {
                uint4  u  = Tq[q];
                float4 w0 = Wq[2 * q], w1 = Wq[2 * q + 1];
                s += bflo(u.x) * w0.x + bfhi(u.x) * w0.y;
                s += bflo(u.y) * w0.z + bfhi(u.y) * w0.w;
                s += bflo(u.z) * w1.x + bfhi(u.z) * w1.y;
                s += bflo(u.w) * w1.z + bfhi(u.w) * w1.w;
            }
            out_trans[((r0 + m) * T + t) * TRANS + o] = s;
        }
        __syncthreads();

        // commit new cell state + next drive (merged phase, one barrier)
        #pragma unroll
        for (int e = 0; e < (ROWS * CS) / NTH; ++e) {
            int idx = e * NTH + tid;
            float v = sm.cnew[idx];
            sm.cf[idx]  = v;
            sm.cbf[idx] = f2bf(v);
        }
        if (tid < ROWS * DRIVE && t + 1 < T) sm.drv[tid] = drv_next;
        __syncthreads();
    }

    // final cell state
    for (int idx = tid; idx < ROWS * CS; idx += NTH) {
        int m = idx >> 5, j = idx & 31;
        out_cell[(r0 + m) * CS + j] = sm.cf[idx];
    }
}

extern "C" void kernel_launch(void* const* d_in, const int* in_sizes, int n_in,
                              void* d_out, int out_size, void* d_ws, size_t ws_size,
                              hipStream_t stream) {
    (void)in_sizes; (void)n_in; (void)out_size; (void)d_ws; (void)ws_size;
    const float* drive = (const float*)d_in[0];
    const float* cell  = (const float*)d_in[1];
    const float* W_in  = (const float*)d_in[2];
    const float* b_in  = (const float*)d_in[3];
    const float* W_td  = (const float*)d_in[4];
    const float* b_td  = (const float*)d_in[5];
    const float* W_ts  = (const float*)d_in[6];
    const float* b_ts  = (const float*)d_in[7];
    const float* W_ud  = (const float*)d_in[8];
    const float* b_ud  = (const float*)d_in[9];
    const float* W_ut  = (const float*)d_in[10];
    const float* b_ut  = (const float*)d_in[11];
    const float* W_out = (const float*)d_in[12];
    const float* b_out = (const float*)d_in[13];

    float* out_trans = (float*)d_out;                         // [B,T,TRANS]
    float* out_cell  = (float*)d_out + (size_t)B * T * TRANS; // [B,CS]

    quantum_cell_kernel<<<NBLK, NTH, 0, stream>>>(
        drive, cell, W_in, b_in, W_td, b_td, W_ts, b_ts,
        W_ud, b_ud, W_ut, b_ut, W_out, b_out, out_trans, out_cell);
}
